// Model_91070486545213
// MI455X (gfx1250) — compile-verified, matrix operand order checked by reading, not run on hardware
//
#include <hip/hip_runtime.h>
#include <hip/hip_bf16.h>
#include <math.h>

typedef _Float16 h16;
typedef __attribute__((ext_vector_type(16))) _Float16 v16h;
typedef __attribute__((ext_vector_type(8)))  float    v8f;

// CDNA5 async global->LDS path (guarded: falls back to gather staging if the
// toolchain doesn't declare the builtins)
#if defined(__HIP_DEVICE_COMPILE__) && \
    __has_builtin(__builtin_amdgcn_global_load_async_to_lds_b128) && \
    __has_builtin(__builtin_amdgcn_s_wait_asynccnt)
#define HAVE_ASYNC_LDS 1
#define AS1 __attribute__((address_space(1)))
#define AS3 __attribute__((address_space(3)))
typedef int v4i_ld __attribute__((vector_size(16)));
#define ASYNC_COPY_B128(gp, lp)                                          \
  __builtin_amdgcn_global_load_async_to_lds_b128(                        \
      (AS1 v4i_ld*)(AS1 h16*)(h16*)(gp), (AS3 v4i_ld*)(AS3 h16*)(lp), 0, 0)
#else
#define HAVE_ASYNC_LDS 0
#endif

// ---------------------------------------------------------------------------
// d_in index map (x first, then params pytree leaves with sorted dict keys)
// ---------------------------------------------------------------------------
enum {
  IN_X = 0,
  DEC_C0B = 1, DEC_C0W = 2, DEC_T1B = 3, DEC_T1W = 4, DEC_T2B = 5, DEC_T2W = 6,
  DEC_T3B = 7, DEC_T3W = 8, DEC_T4B = 9, DEC_T4W = 10, DEC_T5B = 11, DEC_T5W = 12,
  EMB = 13,
  STEM_B = 84, STEM_BB = 85, STEM_G = 86, STEM_W = 87,
  FC1B = 88, FC1W = 89, FC2B = 90, FC2W = 91
};

#define RECON_OFF 1
#define RECON_N   1048576
#define PERP_OFF  1048577
#define CLS_OFF   1048578

// ---------------------------------------------------------------------------
// small helpers
// ---------------------------------------------------------------------------
__global__ void k_f32_to_f16(const float* __restrict__ in, h16* __restrict__ out, int n) {
  int i = blockIdx.x * 256 + threadIdx.x;
  if (i < n) out[i] = (h16)in[i];
}

__global__ void k_recon_out(const h16* __restrict__ in, float* __restrict__ out, int n) {
  int i = blockIdx.x * 256 + threadIdx.x;
  if (i < n) out[i] = (float)in[i];
}

__global__ void k_zero_vq(float* hist, float* lossA) {
  int i = blockIdx.x * 256 + threadIdx.x;
  if (i < 512) hist[i] = 0.f;
  else if (i == 512) lossA[0] = 0.f;
}

// conv weights OIHW f32 -> [Co][Ktp] fp16, Kidx = (ky*kw+kx)*Ci + ci, zero-padded
__global__ void reformat_conv_w(const float* __restrict__ w, h16* __restrict__ w2,
                                int Co, int Ci, int kh, int kw, int Kt, int Ktp) {
  int i = blockIdx.x * 256 + threadIdx.x;
  if (i >= Co * Ktp) return;
  int co = i / Ktp, kidx = i % Ktp;
  float v = 0.f;
  if (kidx < Kt) {
    int ci = kidx % Ci;
    int t  = kidx / Ci;
    int kx = t % kw, ky = t / kw;
    v = w[(((size_t)co * Ci + ci) * kh + ky) * kw + kx];
  }
  w2[i] = (h16)v;
}

// convT weights [Ci][Co][4][4] f32 -> [4 parity classes][Co][4*Ci] fp16
__global__ void reformat_convt_w(const float* __restrict__ w, h16* __restrict__ wpc,
                                 int Ci, int Co) {
  int Kt = 4 * Ci;
  int total = 4 * Co * Kt;
  int i = blockIdx.x * 256 + threadIdx.x;
  if (i >= total) return;
  int pc = i / (Co * Kt);
  int r  = i % (Co * Kt);
  int co = r / Kt, kidx = r % Kt;
  int ci = kidx % Ci, tap = kidx / Ci;
  int pcy = pc >> 1, pcx = pc & 1;
  int ky = (1 - pcy) + 2 * (tap >> 1);
  int kx = (1 - pcx) + 2 * (tap & 1);
  wpc[i] = (h16)w[(((size_t)ci * Co + co) * 4 + ky) * 4 + kx];
}

// ---------------------------------------------------------------------------
// implicit-GEMM convolution with WMMA.  NHWC in/out, weights [Co][Ktp] fp16.
// Block = 128 threads (4 waves).  Tile M=32 output pixels x N=64 channels,
// K staged 64 at a time in LDS -> 4 WMMAs per wave per staging round.
// When Ci % 64 == 0 the A tile rows are contiguous channel runs and are
// staged via GLOBAL_LOAD_ASYNC_TO_LDS_B128 (ASYNCcnt), OOB segments zeroed
// with plain LDS stores by the same owner thread.
// ---------------------------------------------------------------------------
__global__ void __launch_bounds__(128)
conv_wmma_kernel(const h16* __restrict__ in, const h16* __restrict__ w2,
                 const float* __restrict__ bias, h16* __restrict__ out,
                 int Bn, int Hi, int Wi, int Ci, int Co,
                 int kh, int kw, int stride, int pad,
                 int Ho, int Wo, int Kt, int Ktp) {
  __shared__ __align__(128) h16 As[32][64];
  const int tid  = threadIdx.x;
  const int wv   = tid >> 5, lane = tid & 31;
  const int li   = lane & 15, hs = lane >> 4;
  const int Npix = Bn * Ho * Wo;
  const int pix0 = blockIdx.x * 32;
  const int n0   = blockIdx.y * 64 + wv * 16;
  const bool nOK = (n0 + li) < Co;

  v8f acc0 = {}, acc1 = {};
  for (int k0 = 0; k0 < Ktp; k0 += 64) {
#if HAVE_ASYNC_LDS
    if ((Ci & 63) == 0) {
      // 256 16-byte segments: s -> row m = s>>3, seg = s&7 (8 halves each)
      for (int s = tid; s < 256; s += 128) {
        int m = s >> 3, seg = s & 7;
        int kg0 = k0 + seg * 8;
        int pix = pix0 + m;
        bool ok = false;
        const h16* gp = in;
        if (pix < Npix && kg0 < Kt) {
          int ci0 = kg0 % Ci;
          int t   = kg0 / Ci;
          int kx = t % kw, ky = t / kw;
          int b  = pix / (Ho * Wo);
          int r  = pix - b * Ho * Wo;
          int oy = r / Wo, ox = r - oy * Wo;
          int iy = oy * stride - pad + ky;
          int ix = ox * stride - pad + kx;
          if (iy >= 0 && iy < Hi && ix >= 0 && ix < Wi) {
            ok = true;
            gp = &in[((size_t)(b * Hi + iy) * Wi + ix) * Ci + ci0];
          }
        }
        h16* lp = &As[m][seg * 8];
        if (ok) {
          ASYNC_COPY_B128(gp, lp);
        } else {
          uint4 z4 = {0u, 0u, 0u, 0u};
          *(uint4*)lp = z4;
        }
      }
      __builtin_amdgcn_s_wait_asynccnt(0);
    } else
#endif
    {
      for (int e = tid; e < 32 * 64; e += 128) {
        int m = e >> 6, kk = e & 63;
        int kg = k0 + kk;
        int pix = pix0 + m;
        float v = 0.f;
        if (kg < Kt && pix < Npix) {
          int ci = kg % Ci;
          int t  = kg / Ci;
          int kx = t % kw, ky = t / kw;
          int b  = pix / (Ho * Wo);
          int r  = pix - b * Ho * Wo;
          int oy = r / Wo, ox = r - oy * Wo;
          int iy = oy * stride - pad + ky;
          int ix = ox * stride - pad + kx;
          if (iy >= 0 && iy < Hi && ix >= 0 && ix < Wi)
            v = (float)in[((size_t)(b * Hi + iy) * Wi + ix) * Ci + ci];
        }
        As[m][kk] = (h16)v;
      }
    }
    __syncthreads();
#pragma unroll
    for (int k1 = 0; k1 < 64; k1 += 32) {
      v16h bf = {};
      if (nOK) {
        const h16* bp = &w2[(size_t)(n0 + li) * Ktp + k0 + k1 + hs * 16];
        bf = *(const v16h*)bp;
        if (k0 + 128 < Ktp) __builtin_prefetch(bp + 128, 0, 0);
      }
      v16h a0 = *(const v16h*)&As[li][k1 + hs * 16];
      v16h a1 = *(const v16h*)&As[16 + li][k1 + hs * 16];
      acc0 = __builtin_amdgcn_wmma_f32_16x16x32_f16(false, a0, false, bf, (short)0, acc0,
                                                    false, false);
      acc1 = __builtin_amdgcn_wmma_f32_16x16x32_f16(false, a1, false, bf, (short)0, acc1,
                                                    false, false);
    }
    __syncthreads();
  }
  int n = n0 + li;
  if (n < Co) {
    float bs = bias ? bias[n] : 0.f;
    for (int r = 0; r < 8; ++r) {
      int m = r + 8 * hs;
      int p0 = pix0 + m, p1 = pix0 + 16 + m;
      if (p0 < Npix) out[(size_t)p0 * Co + n] = (h16)(acc0[r] + bs);
      if (p1 < Npix) out[(size_t)p1 * Co + n] = (h16)(acc1[r] + bs);
    }
  }
}

// ---------------------------------------------------------------------------
// ConvTranspose2d(k=4,s=2,p=1) as implicit GEMM, tiled by output parity class.
// weights pre-split per class: [4][Co][Kt=4*Ci] fp16.  Same M=32/N=64/K=64
// tiling as conv; async LDS staging when Ci % 64 == 0.
// ---------------------------------------------------------------------------
__global__ void __launch_bounds__(128)
convt_wmma_kernel(const h16* __restrict__ in, const h16* __restrict__ wpc,
                  const float* __restrict__ bias, h16* __restrict__ out,
                  int Bn, int Hi, int Wi, int Ci, int Co) {
  const int Ho = Hi * 2, Wo = Wi * 2, Kt = 4 * Ci;
  const int pc = blockIdx.z, pcy = pc >> 1, pcx = pc & 1;
  const int hh = Hi, wh = Wi;  // per-class output grid == input grid
  const int NpixC = Bn * hh * wh;
  __shared__ __align__(128) h16 As[32][64];
  const int tid = threadIdx.x;
  const int wv = tid >> 5, lane = tid & 31;
  const int li = lane & 15, hs = lane >> 4;
  const int pix0 = blockIdx.x * 32;
  const int n0 = blockIdx.y * 64 + wv * 16;
  const bool nOK = (n0 + li) < Co;
  const h16* wb = wpc + (size_t)pc * Co * Kt;

  v8f acc0 = {}, acc1 = {};
  for (int k0 = 0; k0 < Kt; k0 += 64) {
#if HAVE_ASYNC_LDS
    if ((Ci & 63) == 0) {
      for (int s = tid; s < 256; s += 128) {
        int m = s >> 3, seg = s & 7;
        int kg0 = k0 + seg * 8;
        int pixc = pix0 + m;
        bool ok = false;
        const h16* gp = in;
        if (pixc < NpixC) {
          int ci0 = kg0 % Ci;
          int tap = kg0 / Ci;
          int kySel = tap >> 1, kxSel = tap & 1;
          int b = pixc / (hh * wh);
          int r = pixc - b * hh * wh;
          int oyh = r / wh, oxh = r - oyh * wh;
          int iy = oyh + pcy - kySel;
          int ix = oxh + pcx - kxSel;
          if (iy >= 0 && iy < Hi && ix >= 0 && ix < Wi) {
            ok = true;
            gp = &in[((size_t)(b * Hi + iy) * Wi + ix) * Ci + ci0];
          }
        }
        h16* lp = &As[m][seg * 8];
        if (ok) {
          ASYNC_COPY_B128(gp, lp);
        } else {
          uint4 z4 = {0u, 0u, 0u, 0u};
          *(uint4*)lp = z4;
        }
      }
      __builtin_amdgcn_s_wait_asynccnt(0);
    } else
#endif
    {
      for (int e = tid; e < 32 * 64; e += 128) {
        int m = e >> 6, kk = e & 63;
        int kg = k0 + kk;
        int pixc = pix0 + m;
        float v = 0.f;
        if (pixc < NpixC) {
          int ci = kg % Ci;
          int tap = kg / Ci;
          int kySel = tap >> 1, kxSel = tap & 1;
          int b = pixc / (hh * wh);
          int r = pixc - b * hh * wh;
          int oyh = r / wh, oxh = r - oyh * wh;
          int iy = oyh + pcy - kySel;
          int ix = oxh + pcx - kxSel;
          if (iy >= 0 && iy < Hi && ix >= 0 && ix < Wi)
            v = (float)in[((size_t)(b * Hi + iy) * Wi + ix) * Ci + ci];
        }
        As[m][kk] = (h16)v;
      }
    }
    __syncthreads();
#pragma unroll
    for (int k1 = 0; k1 < 64; k1 += 32) {
      v16h bf = {};
      if (nOK) bf = *(const v16h*)&wb[(size_t)(n0 + li) * Kt + k0 + k1 + hs * 16];
      v16h a0 = *(const v16h*)&As[li][k1 + hs * 16];
      v16h a1 = *(const v16h*)&As[16 + li][k1 + hs * 16];
      acc0 = __builtin_amdgcn_wmma_f32_16x16x32_f16(false, a0, false, bf, (short)0, acc0,
                                                    false, false);
      acc1 = __builtin_amdgcn_wmma_f32_16x16x32_f16(false, a1, false, bf, (short)0, acc1,
                                                    false, false);
    }
    __syncthreads();
  }
  int n = n0 + li;
  if (n < Co) {
    float bs = bias ? bias[n] : 0.f;
    for (int r = 0; r < 8; ++r) {
      int m = r + 8 * hs;
      for (int half = 0; half < 2; ++half) {
        int pixc = pix0 + half * 16 + m;
        if (pixc < NpixC) {
          int b = pixc / (hh * wh);
          int rr = pixc - b * hh * wh;
          int oyh = rr / wh, oxh = rr - oyh * wh;
          int oy = 2 * oyh + pcy, ox = 2 * oxh + pcx;
          float v = (half ? acc1[r] : acc0[r]) + bs;
          out[((size_t)(b * Ho + oy) * Wo + ox) * Co + n] = (h16)v;
        }
      }
    }
  }
}

// ---------------------------------------------------------------------------
// plain WMMA GEMM: C[M,N] f32 = A[M,K]h * B'[N,K]h  (B stored N-major)
// M=32 x N=64 tile per block, K unrolled by 64 -> 4 WMMAs per iteration.
// used for VQ distances (M=1024, N=512, K=512) -- all aligned, no guards
// ---------------------------------------------------------------------------
__global__ void __launch_bounds__(128)
gemm_wmma_kernel(const h16* __restrict__ A, const h16* __restrict__ Bm,
                 float* __restrict__ C, int M, int N, int K) {
  const int wv = threadIdx.x >> 5, lane = threadIdx.x & 31;
  const int li = lane & 15, hs = lane >> 4;
  const int m0 = blockIdx.x * 32, n0 = blockIdx.y * 64 + wv * 16;
  v8f acc0 = {}, acc1 = {};
  for (int k0 = 0; k0 < K; k0 += 64) {
#pragma unroll
    for (int k1 = 0; k1 < 64; k1 += 32) {
      const h16* ap0 = &A[(size_t)(m0 + li) * K + k0 + k1 + hs * 16];
      const h16* ap1 = &A[(size_t)(m0 + 16 + li) * K + k0 + k1 + hs * 16];
      const h16* bp  = &Bm[(size_t)(n0 + li) * K + k0 + k1 + hs * 16];
      if (k0 + 128 < K) { __builtin_prefetch(ap0 + 128, 0, 0); __builtin_prefetch(bp + 128, 0, 0); }
      v16h a0 = *(const v16h*)ap0;
      v16h a1 = *(const v16h*)ap1;
      v16h b  = *(const v16h*)bp;
      acc0 = __builtin_amdgcn_wmma_f32_16x16x32_f16(false, a0, false, b, (short)0, acc0,
                                                    false, false);
      acc1 = __builtin_amdgcn_wmma_f32_16x16x32_f16(false, a1, false, b, (short)0, acc1,
                                                    false, false);
    }
  }
  int n = n0 + li;
  for (int r = 0; r < 8; ++r) {
    int m = m0 + r + 8 * hs;
    C[(size_t)m * N + n] = acc0[r];
    C[(size_t)(m + 16) * N + n] = acc1[r];
  }
}

// ---------------------------------------------------------------------------
// BatchNorm (training mode): per-channel stats over all pixels, then apply.
// ---------------------------------------------------------------------------
__global__ void bn_stats_kernel(const h16* __restrict__ x, int Npix, int C,
                                float* __restrict__ mean, float* __restrict__ var) {
  __shared__ float s1[256], s2[256];
  int c = blockIdx.x;
  float a = 0.f, b = 0.f;
  for (int p = threadIdx.x; p < Npix; p += 256) {
    float v = (float)x[(size_t)p * C + c];
    a += v; b += v * v;
  }
  s1[threadIdx.x] = a; s2[threadIdx.x] = b;
  __syncthreads();
  for (int s = 128; s > 0; s >>= 1) {
    if (threadIdx.x < s) { s1[threadIdx.x] += s1[threadIdx.x + s]; s2[threadIdx.x] += s2[threadIdx.x + s]; }
    __syncthreads();
  }
  if (threadIdx.x == 0) {
    float m = s1[0] / (float)Npix;
    mean[c] = m;
    var[c]  = s2[0] / (float)Npix - m * m;
  }
}

__global__ void bn_apply_kernel(const h16* __restrict__ x, h16* __restrict__ y,
                                const float* __restrict__ g, const float* __restrict__ b,
                                const float* __restrict__ mean, const float* __restrict__ var,
                                const h16* __restrict__ res, int relu, int total, int C) {
  int i = blockIdx.x * 256 + threadIdx.x;
  if (i >= total) return;
  int c = i % C;
  float v = ((float)x[i] - mean[c]) * rsqrtf(var[c] + 1e-5f) * g[c] + b[c];
  if (res) v += (float)res[i];
  if (relu) v = fmaxf(v, 0.f);
  y[i] = (h16)v;
}

__global__ void maxpool_kernel(const h16* __restrict__ in, h16* __restrict__ out,
                               int Bn, int Hi, int Wi, int C) {
  int Ho = Hi / 2, Wo = Wi / 2;
  int total = Bn * Ho * Wo * C;
  int i = blockIdx.x * 256 + threadIdx.x;
  if (i >= total) return;
  int c = i % C;
  int r = i / C;
  int ox = r % Wo; r /= Wo;
  int oy = r % Ho;
  int b = r / Ho;
  float m = -1e30f;
  for (int ky = 0; ky < 3; ++ky)
    for (int kx = 0; kx < 3; ++kx) {
      int iy = oy * 2 - 1 + ky, ix = ox * 2 - 1 + kx;
      if (iy >= 0 && iy < Hi && ix >= 0 && ix < Wi)
        m = fmaxf(m, (float)in[((size_t)(b * Hi + iy) * Wi + ix) * C + c]);
    }
  out[i] = (h16)m;
}

// ---------------------------------------------------------------------------
// VQ: argmin over codes, gather quantized rows, loss + histogram
// ---------------------------------------------------------------------------
__global__ void k_enorm(const float* __restrict__ E, float* __restrict__ Enorm) {
  int k = blockIdx.x * 256 + threadIdx.x;
  if (k >= 512) return;
  float s = 0.f;
  for (int d = 0; d < 512; ++d) { float e = E[(size_t)k * 512 + d]; s += e * e; }
  Enorm[k] = s;
}

__global__ void vq_kernel(const float* __restrict__ dot, const float* __restrict__ Enorm,
                          const float* __restrict__ E, const h16* __restrict__ z,
                          h16* __restrict__ q, float* __restrict__ lossA,
                          float* __restrict__ hist) {
  __shared__ float smin[256];
  __shared__ int   sidx[256];
  int n = blockIdx.x;
  float best = 1e30f; int bidx = 0;
  for (int k = threadIdx.x; k < 512; k += 256) {
    float d = Enorm[k] - 2.f * dot[(size_t)n * 512 + k];
    if (d < best) { best = d; bidx = k; }
  }
  smin[threadIdx.x] = best; sidx[threadIdx.x] = bidx;
  __syncthreads();
  for (int s = 128; s > 0; s >>= 1) {
    if (threadIdx.x < s) {
      float ov = smin[threadIdx.x + s]; int oi = sidx[threadIdx.x + s];
      if (ov < smin[threadIdx.x] || (ov == smin[threadIdx.x] && oi < sidx[threadIdx.x])) {
        smin[threadIdx.x] = ov; sidx[threadIdx.x] = oi;
      }
    }
    __syncthreads();
  }
  int idx = sidx[0];
  __syncthreads();
  float ls = 0.f;
  for (int d = threadIdx.x; d < 512; d += 256) {
    float e = E[(size_t)idx * 512 + d];
    float zz = (float)z[(size_t)n * 512 + d];
    q[(size_t)n * 512 + d] = (h16)e;
    float df = e - zz;
    ls += df * df;
  }
  smin[threadIdx.x] = ls;
  __syncthreads();
  for (int s = 128; s > 0; s >>= 1) {
    if (threadIdx.x < s) smin[threadIdx.x] += smin[threadIdx.x + s];
    __syncthreads();
  }
  if (threadIdx.x == 0) {
    atomicAdd(lossA, smin[0]);
    atomicAdd(&hist[idx], 1.0f);
  }
}

__global__ void finalize_kernel(const float* __restrict__ lossA,
                                const float* __restrict__ hist, float* __restrict__ out) {
  __shared__ float sh[256];
  float s = 0.f;
  for (int k = threadIdx.x; k < 512; k += 256) {
    float avg = hist[k] * (1.f / 1024.f);
    s += avg * logf(avg + 1e-10f);
  }
  sh[threadIdx.x] = s;
  __syncthreads();
  for (int st = 128; st > 0; st >>= 1) {
    if (threadIdx.x < st) sh[threadIdx.x] += sh[threadIdx.x + st];
    __syncthreads();
  }
  if (threadIdx.x == 0) {
    out[0] = 1.25f * lossA[0] / (1024.f * 512.f);   // q_loss + 0.25*e_loss, both == MSE
    out[PERP_OFF] = expf(-sh[0]);
  }
}

// ---------------------------------------------------------------------------
// classifier head: flat[b, c*16+pix] = q[b, pix, c]  (q NHWC [64,16,512])
// ---------------------------------------------------------------------------
__global__ void fc1_kernel(const h16* __restrict__ q, const float* __restrict__ w,
                           const float* __restrict__ b, float* __restrict__ hbuf) {
  __shared__ float sh[256];
  int o = blockIdx.x;   // 0..255
  int bi = blockIdx.y;  // 0..63
  float s = 0.f;
  for (int i = threadIdx.x; i < 8192; i += 256) {
    int c = i >> 4, pix = i & 15;
    s += (float)q[((size_t)bi * 16 + pix) * 512 + c] * w[(size_t)o * 8192 + i];
  }
  sh[threadIdx.x] = s;
  __syncthreads();
  for (int st = 128; st > 0; st >>= 1) {
    if (threadIdx.x < st) sh[threadIdx.x] += sh[threadIdx.x + st];
    __syncthreads();
  }
  if (threadIdx.x == 0) hbuf[(size_t)bi * 256 + o] = fmaxf(sh[0] + b[o], 0.f);
}

__global__ void fc2_kernel(const float* __restrict__ hbuf, const float* __restrict__ w,
                           const float* __restrict__ b, float* __restrict__ cls) {
  __shared__ float sh[256];
  int bi = blockIdx.x;
  float s = hbuf[(size_t)bi * 256 + threadIdx.x] * w[threadIdx.x];
  sh[threadIdx.x] = s;
  __syncthreads();
  for (int st = 128; st > 0; st >>= 1) {
    if (threadIdx.x < st) sh[threadIdx.x] += sh[threadIdx.x + st];
    __syncthreads();
  }
  if (threadIdx.x == 0) cls[bi] = 1.f / (1.f + expf(-(sh[0] + b[0])));
}

// ---------------------------------------------------------------------------
// host-side orchestration helpers
// ---------------------------------------------------------------------------
static void run_conv(hipStream_t s, const h16* in, const float* wf, const float* bias,
                     h16* wsc, h16* out, int Bn, int Hi, int Wi, int Ci, int Co,
                     int k, int stride, int pad, int Ho, int Wo) {
  int Kt = k * k * Ci;
  int Ktp = (Kt + 63) & ~63;
  int tot = Co * Ktp;
  reformat_conv_w<<<(tot + 255) / 256, 256, 0, s>>>(wf, wsc, Co, Ci, k, k, Kt, Ktp);
  int Npix = Bn * Ho * Wo;
  dim3 g((Npix + 31) / 32, (Co + 63) / 64);
  conv_wmma_kernel<<<g, 128, 0, s>>>(in, wsc, bias, out, Bn, Hi, Wi, Ci, Co,
                                     k, k, stride, pad, Ho, Wo, Kt, Ktp);
}

static void run_convt(hipStream_t s, const h16* in, const float* wf, const float* bias,
                      h16* wsc, h16* out, int Bn, int Hi, int Wi, int Ci, int Co) {
  int Kt = 4 * Ci;
  int tot = 4 * Co * Kt;
  reformat_convt_w<<<(tot + 255) / 256, 256, 0, s>>>(wf, wsc, Ci, Co);
  int NpixC = Bn * Hi * Wi;
  dim3 g((NpixC + 31) / 32, (Co + 63) / 64, 4);
  convt_wmma_kernel<<<g, 128, 0, s>>>(in, wsc, bias, out, Bn, Hi, Wi, Ci, Co);
}

static void run_bn(hipStream_t s, h16* x, const float* g, const float* b,
                   float* mean, float* var, int Npix, int C, const h16* res, bool relu) {
  bn_stats_kernel<<<C, 256, 0, s>>>(x, Npix, C, mean, var);
  int tot = Npix * C;
  bn_apply_kernel<<<(tot + 255) / 256, 256, 0, s>>>(x, x, g, b, mean, var, res,
                                                    relu ? 1 : 0, tot, C);
}

struct RBIdx { int w1, b1, g1, bb1, w2, b2, g2, bb2, w3, b3; };

static void resblock(hipStream_t s, void* const* d_in, const RBIdx& P,
                     h16* X, h16* Y, h16* Z, h16* wsc, float* meanB, float* varB,
                     int Bn, int Hin, int Cin, int Cout, int stride) {
  int Hout = Hin / stride;
  run_conv(s, X, (const float*)d_in[P.w1], (const float*)d_in[P.b1], wsc, Y,
           Bn, Hin, Hin, Cin, Cout, 3, stride, 1, Hout, Hout);
  run_bn(s, Y, (const float*)d_in[P.g1], (const float*)d_in[P.bb1], meanB, varB,
         Bn * Hout * Hout, Cout, nullptr, true);
  run_conv(s, Y, (const float*)d_in[P.w2], (const float*)d_in[P.b2], wsc, Z,
           Bn, Hout, Hout, Cout, Cout, 3, 1, 1, Hout, Hout);
  const h16* res = X;
  if (P.w3 >= 0) {
    run_conv(s, X, (const float*)d_in[P.w3], (const float*)d_in[P.b3], wsc, Y,
             Bn, Hin, Hin, Cin, Cout, 1, stride, 0, Hout, Hout);
    res = Y;
  }
  run_bn(s, Z, (const float*)d_in[P.g2], (const float*)d_in[P.bb2], meanB, varB,
         Bn * Hout * Hout, Cout, res, true);
}

// ---------------------------------------------------------------------------
extern "C" void kernel_launch(void* const* d_in, const int* in_sizes, int n_in,
                              void* d_out, int out_size, void* d_ws, size_t ws_size,
                              hipStream_t stream) {
  (void)in_sizes; (void)n_in; (void)out_size; (void)ws_size;
  float* out = (float*)d_out;
  auto F = [&](int i) { return (const float*)d_in[i]; };

  char* base = (char*)d_ws;
  size_t off = 0;
  auto alloc = [&](size_t bytes) -> void* {
    void* p = base + off;
    off = (off + bytes + 255) & ~(size_t)255;
    return p;
  };
  const size_t BIGE = 16777216;  // 64*64*64*64 elems, largest activation
  h16* bufA = (h16*)alloc(BIGE * 2);
  h16* bufB = (h16*)alloc(BIGE * 2);
  h16* bufC = (h16*)alloc(BIGE * 2);
  h16* xh   = (h16*)alloc((size_t)1048576 * 2);
  h16* wsc  = (h16*)alloc((size_t)4194304 * 2);
  h16* Eh   = (h16*)alloc((size_t)262144 * 2);
  h16* q    = (h16*)alloc((size_t)524288 * 2);
  float* dist  = (float*)alloc((size_t)524288 * 4);
  float* Enorm = (float*)alloc(512 * 4);
  float* hist  = (float*)alloc(512 * 4);
  float* lossA = (float*)alloc(4);
  float* meanB = (float*)alloc(512 * 4);
  float* varB  = (float*)alloc(512 * 4);
  float* hbuf  = (float*)alloc((size_t)16384 * 4);

  // ---- input to fp16 (NCHW with C=1 == NHWC) ----
  k_f32_to_f16<<<(1048576 + 255) / 256, 256, 0, stream>>>(F(IN_X), xh, 1048576);

  // ---- encoder: stem conv 7x7 s2 + BN + ReLU + maxpool 3x3 s2 ----
  run_conv(stream, xh, F(STEM_W), F(STEM_B), wsc, bufA, 64, 128, 128, 1, 64, 7, 2, 3, 64, 64);
  run_bn(stream, bufA, F(STEM_G), F(STEM_BB), meanB, varB, 64 * 64 * 64, 64, nullptr, true);
  {
    int tot = 64 * 32 * 32 * 64;
    maxpool_kernel<<<(tot + 255) / 256, 256, 0, stream>>>(bufA, bufB, 64, 64, 64, 64);
  }

  // ---- residual stages ----
  struct BlockCfg { RBIdx p; int Hin, Cin, Cout, stride; };
  const BlockCfg blocks[8] = {
    {{20, 14, 18, 16, 21, 15, 19, 17, -1, -1}, 32,  64,  64, 1},
    {{28, 22, 26, 24, 29, 23, 27, 25, -1, -1}, 32,  64,  64, 1},
    {{37, 30, 35, 33, 38, 31, 36, 34, 39, 32}, 32,  64, 128, 2},
    {{46, 40, 44, 42, 47, 41, 45, 43, -1, -1}, 16, 128, 128, 1},
    {{55, 48, 53, 51, 56, 49, 54, 52, 57, 50}, 16, 128, 256, 2},
    {{64, 58, 62, 60, 65, 59, 63, 61, -1, -1},  8, 256, 256, 1},
    {{73, 66, 71, 69, 74, 67, 72, 70, 75, 68},  8, 256, 512, 2},
    {{82, 76, 80, 78, 83, 77, 81, 79, -1, -1},  4, 512, 512, 1},
  };
  h16 *X = bufB, *Y = bufC, *Z = bufA;
  for (int i = 0; i < 8; ++i) {
    const BlockCfg& c = blocks[i];
    resblock(stream, d_in, c.p, X, Y, Z, wsc, meanB, varB, 64, c.Hin, c.Cin, c.Cout, c.stride);
    h16* t = X; X = Z; Z = Y; Y = t;  // output becomes next input
  }
  // X = z: NHWC [64,4,4,512] == [1024,512]

  // ---- VQ: WMMA GEMM z @ E^T, argmin, gather, loss, perplexity ----
  k_f32_to_f16<<<(262144 + 255) / 256, 256, 0, stream>>>(F(EMB), Eh, 262144);
  {
    dim3 gg(1024 / 32, 512 / 64);
    gemm_wmma_kernel<<<gg, 128, 0, stream>>>(X, Eh, dist, 1024, 512, 512);
  }
  k_enorm<<<2, 256, 0, stream>>>(F(EMB), Enorm);
  k_zero_vq<<<3, 256, 0, stream>>>(hist, lossA);
  vq_kernel<<<1024, 256, 0, stream>>>(dist, Enorm, F(EMB), X, q, lossA, hist);
  finalize_kernel<<<1, 256, 0, stream>>>(lossA, hist, out);

  // ---- classifier head (uses straight-through q, forward value == q) ----
  {
    dim3 gfc(256, 64);
    fc1_kernel<<<gfc, 256, 0, stream>>>(q, F(FC1W), F(FC1B), hbuf);
    fc2_kernel<<<64, 256, 0, stream>>>(hbuf, F(FC2W), F(FC2B), out + CLS_OFF);
  }

  // ---- decoder: 1x1 conv + 5x ConvTranspose(k4,s2,p1) ----
  run_conv(stream, q, F(DEC_C0W), F(DEC_C0B), wsc, bufB, 64, 4, 4, 512, 512, 1, 1, 0, 4, 4);
  run_convt(stream, bufB, F(DEC_T1W), F(DEC_T1B), wsc, bufC, 64,  4,  4, 512, 256);
  run_convt(stream, bufC, F(DEC_T2W), F(DEC_T2B), wsc, bufA, 64,  8,  8, 256, 128);
  run_convt(stream, bufA, F(DEC_T3W), F(DEC_T3B), wsc, bufB, 64, 16, 16, 128,  64);
  run_convt(stream, bufB, F(DEC_T4W), F(DEC_T4B), wsc, bufC, 64, 32, 32,  64,  32);
  run_convt(stream, bufC, F(DEC_T5W), F(DEC_T5B), wsc, bufA, 64, 64, 64,  32,   1);
  k_recon_out<<<(RECON_N + 255) / 256, 256, 0, stream>>>(bufA, out + RECON_OFF, RECON_N);
}
